// Triton_69887707840947
// MI455X (gfx1250) — matrix operand order repaired and checked by reading, not compiled, not run on hardware
//
#include <hip/hip_runtime.h>
#include <stdint.h>
#include <stddef.h>

// Problem dims (fixed by the reference)
#define M_TOT  (8 * 8192)   // B*S tokens
#define K_DIM  1024
#define N_DIM  1024

// Tiling
#define BM     128          // workgroup M tile
#define BN     64           // workgroup N tile
#define BK     32           // K slice per stage (one WMMA K)
#define NKB    (K_DIM / BK) // 32 K-slices
#define LDSR   40           // ushorts per LDS row: 32 data + 8 pad = 80 B

typedef __bf16    v16bf __attribute__((ext_vector_type(16)));
typedef float     v8f   __attribute__((ext_vector_type(8)));
typedef float     f32x4 __attribute__((ext_vector_type(4)));
typedef uint32_t  u32x4 __attribute__((ext_vector_type(4)));

struct FragBits { u32x4 lo, hi; };   // 32 bytes == one v16bf fragment

// fp32 -> bf16 via hardware convert (expect v_cvt_pk_bf16_f32 on gfx1250)
__device__ __forceinline__ uint32_t pack_bf16x2(float a, float b) {
  __bf16 x = (__bf16)a;
  __bf16 y = (__bf16)b;
  uint16_t ux = __builtin_bit_cast(uint16_t, x);
  uint16_t uy = __builtin_bit_cast(uint16_t, y);
  return (uint32_t)ux | ((uint32_t)uy << 16);
}

// A-matrix (16-bit 16x32) per-lane layout: lane group g = lane>>4 holds
// K = g*8..g*8+7 (VGPR0-3) and K = 16+g*8..16+g*8+7 (VGPR4-7).
__device__ __forceinline__ v16bf load_frag_A(const uint16_t* rowBase, int g) {
  FragBits f;
  f.lo = *(const u32x4*)(rowBase + g * 8);
  f.hi = *(const u32x4*)(rowBase + 16 + g * 8);
  return __builtin_bit_cast(v16bf, f);
}

// B-matrix (32x16) per-lane layout: lane group g holds K = g*8..g*8+7 and K = 16+g*8..16+g*8+7 (same as A).
__device__ __forceinline__ v16bf load_frag_B(const uint16_t* rowBase, int g) {
  FragBits f;
  f.lo = *(const u32x4*)(rowBase + g * 8);
  f.hi = *(const u32x4*)(rowBase + 16 + g * 8);
  return __builtin_bit_cast(v16bf, f);
}

__device__ __forceinline__ v8f wmma_bf16(v16bf a, v16bf b, v8f c) {
  return __builtin_amdgcn_wmma_f32_16x16x32_bf16(false, a, false, b,
                                                 (short)0, c, false, false);
}

// --- CDNA5 async global->LDS copy (ASYNCcnt-tracked; bypasses VGPRs) ---
// Flat-pointer low 32 bits of a __shared__ object are its LDS byte offset.
__device__ __forceinline__ uint32_t lds_off32(const void* p) {
  return (uint32_t)(uintptr_t)p;
}
__device__ __forceinline__ void async_copy_b128(uint32_t lds_off, const void* gaddr) {
  asm volatile("global_load_async_to_lds_b128 %0, %1, off"
               :: "v"(lds_off), "v"(gaddr) : "memory");
}
__device__ __forceinline__ void wait_async_le4() {
  asm volatile("s_wait_asynccnt 0x4" ::: "memory");
}
__device__ __forceinline__ void wait_async_0() {
  asm volatile("s_wait_asynccnt 0x0" ::: "memory");
}

// ============================================================================
// Prepass: elementwise fp32 -> bf16 (each element converted exactly once)
// ============================================================================
extern "C" __global__ void __launch_bounds__(256)
cvt_f32_to_bf16(const float* __restrict__ src, uint16_t* __restrict__ dst,
                long long n) {
  long long i = ((long long)blockIdx.x * 256 + threadIdx.x) * 8;
  if (i + 8 <= n) {
    f32x4 a = ((const f32x4*)(src + i))[0];
    f32x4 b = ((const f32x4*)(src + i))[1];
    u32x4 o;
    o[0] = pack_bf16x2(a[0], a[1]);
    o[1] = pack_bf16x2(a[2], a[3]);
    o[2] = pack_bf16x2(b[0], b[1]);
    o[3] = pack_bf16x2(b[2], b[3]);
    *(u32x4*)(dst + i) = o;
  }
}

// ============================================================================
// Main GEMM (bf16 inputs from workspace): async double-buffered LDS staging,
// dual accumulators (w_t and w_v), per-row routed epilogue select.
// ============================================================================
extern "C" __global__ void __launch_bounds__(256)
routed_gemm_bf16_async(const uint16_t* __restrict__ xb,
                       const int*      __restrict__ visual_mask,
                       const uint16_t* __restrict__ wvb,
                       const uint16_t* __restrict__ wtb,
                       float*          __restrict__ out)
{
  __shared__ alignas(16) uint16_t ldsA [2][BM * LDSR];  // 2 x 10240 B
  __shared__ alignas(16) uint16_t ldsBt[2][BN * LDSR];  // 2 x  5120 B
  __shared__ alignas(16) uint16_t ldsBv[2][BN * LDSR];  // 2 x  5120 B

  const int tid  = (int)threadIdx.x;
  const int wid  = tid >> 5;          // 8 waves (wave32)
  const int lane = tid & 31;

  const int blockN = (int)blockIdx.x * BN;
  const int blockM = (int)blockIdx.y * BM;

  // waves 4(M) x 2(N); each wave owns a 32x32 output tile = 2x2 WMMA tiles
  const int wm = (wid >> 1) * 32;
  const int wn = (wid & 1) * 32;

  // fragment lane decomposition
  const int fr = lane & 15;
  const int g  = lane >> 4;

  // ---- per-thread async-copy assignment (16 B strips of BK=32 bf16 rows) ----
  // A tile: 128 rows x 4 strips = 512 strips -> strips tid and tid+256
  // B tiles: 64 rows x 4 strips = 256 strips each -> strip tid for w_t and w_v
  const int sA0r = tid >> 2,          sA0s = tid & 3;
  const int sA1r = (tid + 256) >> 2,  sA1s = tid & 3;   // (tid+256)&3 == tid&3
  const int sBr  = tid >> 2,          sBs  = tid & 3;

  const uint16_t* gA0 = xb  + (size_t)(blockM + sA0r) * K_DIM + sA0s * 8;
  const uint16_t* gA1 = xb  + (size_t)(blockM + sA1r) * K_DIM + sA1s * 8;
  const uint16_t* gBt = wtb + (size_t)(blockN + sBr ) * K_DIM + sBs  * 8;
  const uint16_t* gBv = wvb + (size_t)(blockN + sBr ) * K_DIM + sBs  * 8;

  uint32_t oA0[2], oA1[2], oBt[2], oBv[2];
#pragma unroll
  for (int b = 0; b < 2; ++b) {
    oA0[b] = lds_off32(&ldsA [b][sA0r * LDSR + sA0s * 8]);
    oA1[b] = lds_off32(&ldsA [b][sA1r * LDSR + sA1s * 8]);
    oBt[b] = lds_off32(&ldsBt[b][sBr  * LDSR + sBs  * 8]);
    oBv[b] = lds_off32(&ldsBv[b][sBr  * LDSR + sBs  * 8]);
  }

  auto issue = [&](int buf, int kElem) {   // 4 async instructions per wave
    async_copy_b128(oA0[buf], gA0 + kElem);
    async_copy_b128(oA1[buf], gA1 + kElem);
    async_copy_b128(oBt[buf], gBt + kElem);
    async_copy_b128(oBv[buf], gBv + kElem);
  };

  v8f acc_t[2][2], acc_v[2][2];
#pragma unroll
  for (int i = 0; i < 2; ++i)
#pragma unroll
    for (int j = 0; j < 2; ++j)
#pragma unroll
      for (int v = 0; v < 8; ++v) { acc_t[i][j][v] = 0.0f; acc_v[i][j][v] = 0.0f; }

  issue(0, 0);   // prime buffer 0

#pragma unroll 1
  for (int ik = 0; ik < NKB; ++ik) {
    const int cur = ik & 1;
    const bool more = (ik + 1) < NKB;

    if (more) issue(cur ^ 1, (ik + 1) * BK);   // overlap copy(k+1) with wmma(k)

    // async loads complete FIFO per wave: <=4 outstanding => current buf done
    if (more) wait_async_le4(); else wait_async_0();
    __syncthreads();   // other waves' copies into our fragments are done too

    const uint16_t* A  = ldsA [cur];
    const uint16_t* Bt = ldsBt[cur];
    const uint16_t* Bv = ldsBv[cur];

    const v16bf a0  = load_frag_A(&A [(wm      + fr) * LDSR], g);
    const v16bf a1  = load_frag_A(&A [(wm + 16 + fr) * LDSR], g);
    const v16bf bt0 = load_frag_B(&Bt[(wn      + fr) * LDSR], g);
    const v16bf bt1 = load_frag_B(&Bt[(wn + 16 + fr) * LDSR], g);
    const v16bf bv0 = load_frag_B(&Bv[(wn      + fr) * LDSR], g);
    const v16bf bv1 = load_frag_B(&Bv[(wn + 16 + fr) * LDSR], g);

    acc_t[0][0] = wmma_bf16(a0, bt0, acc_t[0][0]);
    acc_t[0][1] = wmma_bf16(a0, bt1, acc_t[0][1]);
    acc_t[1][0] = wmma_bf16(a1, bt0, acc_t[1][0]);
    acc_t[1][1] = wmma_bf16(a1, bt1, acc_t[1][1]);
    acc_v[0][0] = wmma_bf16(a0, bv0, acc_v[0][0]);
    acc_v[0][1] = wmma_bf16(a0, bv1, acc_v[0][1]);
    acc_v[1][0] = wmma_bf16(a1, bv0, acc_v[1][0]);
    acc_v[1][1] = wmma_bf16(a1, bv1, acc_v[1][1]);

    __syncthreads();   // everyone done reading buf before it is refilled
  }

  // Routed epilogue. C/D layout: lane l -> N = l&15 ; M = vgpr + (l>>4)*8
#pragma unroll
  for (int i = 0; i < 2; ++i) {
    const int m0 = blockM + wm + i * 16 + g * 8;
#pragma unroll
    for (int v = 0; v < 8; ++v) {
      const bool vis = visual_mask[m0 + v] != 0;
#pragma unroll
      for (int j = 0; j < 2; ++j) {
        const int n = blockN + wn + j * 16 + fr;
        out[(size_t)(m0 + v) * N_DIM + n] = vis ? acc_v[i][j][v] : acc_t[i][j][v];
      }
    }
  }
}

// ============================================================================
// Fallback (workspace too small): round-1 style fused-convert GEMM,
// single accumulator via complementary row zeroing over two K phases.
// ============================================================================
__device__ __forceinline__ void stage_strip(uint16_t* dst, const float* src, bool keep) {
  f32x4 f0 = ((const f32x4*)src)[0];
  f32x4 f1 = ((const f32x4*)src)[1];
  f32x4 f2 = ((const f32x4*)src)[2];
  f32x4 f3 = ((const f32x4*)src)[3];
  u32x4 lo, hi;
  lo[0] = pack_bf16x2(f0[0], f0[1]); lo[1] = pack_bf16x2(f0[2], f0[3]);
  lo[2] = pack_bf16x2(f1[0], f1[1]); lo[3] = pack_bf16x2(f1[2], f1[3]);
  hi[0] = pack_bf16x2(f2[0], f2[1]); hi[1] = pack_bf16x2(f2[2], f2[3]);
  hi[2] = pack_bf16x2(f3[0], f3[1]); hi[3] = pack_bf16x2(f3[2], f3[3]);
#pragma unroll
  for (int i = 0; i < 4; ++i) {
    lo[i] = keep ? lo[i] : 0u;
    hi[i] = keep ? hi[i] : 0u;
  }
  ((u32x4*)dst)[0] = lo;
  ((u32x4*)dst)[1] = hi;
}

extern "C" __global__ void __launch_bounds__(256)
routed_dual_linear_fused(const float* __restrict__ x,
                         const int*   __restrict__ visual_mask,
                         const float* __restrict__ w_v,
                         const float* __restrict__ w_t,
                         float*       __restrict__ out)
{
  __shared__ alignas(16) uint16_t ldsA[BM * LDSR];
  __shared__ alignas(16) uint16_t ldsB[BN * LDSR];

  const int tid  = (int)threadIdx.x;
  const int wid  = tid >> 5;
  const int lane = tid & 31;
  const int blockN = (int)blockIdx.x * BN;
  const int blockM = (int)blockIdx.y * BM;
  const int wm = (wid >> 1) * 32;
  const int wn = (wid & 1) * 32;
  const int aRow  = tid >> 1;
  const int chunk = (tid & 1) * 16;
  const int bRow  = (tid & 127) >> 1;
  const int fr = lane & 15;
  const int g  = lane >> 4;
  const int myMask = visual_mask[blockM + aRow];

  const uint16_t* aBase0 = &ldsA[(wm + fr) * LDSR];
  const uint16_t* aBase1 = &ldsA[(wm + 16 + fr) * LDSR];
  const uint16_t* bBase0 = &ldsB[(wn + fr) * LDSR];
  const uint16_t* bBase1 = &ldsB[(wn + 16 + fr) * LDSR];

  v8f acc[2][2];
#pragma unroll
  for (int i = 0; i < 2; ++i)
#pragma unroll
    for (int j = 0; j < 2; ++j)
#pragma unroll
      for (int v = 0; v < 8; ++v) acc[i][j][v] = 0.0f;

#pragma unroll 1
  for (int phase = 0; phase < 2; ++phase) {
    const float* w   = phase ? w_v : w_t;
    const bool  keep = (myMask != 0) == (phase == 1);
#pragma unroll 1
    for (int kb = 0; kb < K_DIM; kb += BK) {
      const float* aSrc = x + (size_t)(blockM + aRow) * K_DIM + (size_t)(kb + chunk);
      stage_strip(&ldsA[aRow * LDSR + chunk], aSrc, keep);
      if (tid < 128) {
        const float* bSrc = w + (size_t)(blockN + bRow) * K_DIM + (size_t)(kb + chunk);
        stage_strip(&ldsB[bRow * LDSR + chunk], bSrc, true);
        if (kb + BK < K_DIM) __builtin_prefetch(bSrc + BK, 0, 0);
      }
      if (kb + BK < K_DIM) __builtin_prefetch(aSrc + BK, 0, 0);
      __syncthreads();

      const v16bf a0 = load_frag_A(aBase0, g);
      const v16bf a1 = load_frag_A(aBase1, g);
      const v16bf b0 = load_frag_B(bBase0, g);
      const v16bf b1 = load_frag_B(bBase1, g);
      acc[0][0] = wmma_bf16(a0, b0, acc[0][0]);
      acc[0][1] = wmma_bf16(a0, b1, acc[0][1]);
      acc[1][0] = wmma_bf16(a1, b0, acc[1][0]);
      acc[1][1] = wmma_bf16(a1, b1, acc[1][1]);
      __syncthreads();
    }
  }

#pragma unroll
  for (int i = 0; i < 2; ++i)
#pragma unroll
    for (int j = 0; j < 2; ++j) {
      const int n  = blockN + wn + j * 16 + fr;
      const int m0 = blockM + wm + i * 16 + g * 8;
      float* o = out + (size_t)m0 * N_DIM + (size_t)n;
#pragma unroll
      for (int v = 0; v < 8; ++v)
        o[(size_t)v * N_DIM] = acc[i][j][v];
    }
}

// ============================================================================
extern "C" void kernel_launch(void* const* d_in, const int* in_sizes, int n_in,
                              void* d_out, int out_size, void* d_ws, size_t ws_size,
                              hipStream_t stream) {
  (void)in_sizes; (void)n_in; (void)out_size;
  const float* x    = (const float*)d_in[0];
  const int*   mask = (const int*)d_in[1];
  const float* w_v  = (const float*)d_in[2];
  const float* w_t  = (const float*)d_in[3];
  float*       out  = (float*)d_out;

  const long long xElems = (long long)M_TOT * K_DIM;   // 67,108,864
  const long long wElems = (long long)N_DIM * K_DIM;   //  1,048,576
  const size_t    need   = (size_t)(xElems + 2 * wElems) * sizeof(uint16_t);

  dim3 grid(N_DIM / BN, M_TOT / BM, 1);   // 16 x 512
  dim3 block(256, 1, 1);

  if (d_ws != nullptr && ws_size >= need) {
    uint16_t* xb  = (uint16_t*)d_ws;
    uint16_t* wtb = xb + xElems;
    uint16_t* wvb = wtb + wElems;
    hipLaunchKernelGGL(cvt_f32_to_bf16, dim3((unsigned)(xElems / 2048)), block, 0, stream,
                       x, xb, xElems);
    hipLaunchKernelGGL(cvt_f32_to_bf16, dim3((unsigned)(wElems / 2048)), block, 0, stream,
                       w_t, wtb, wElems);
    hipLaunchKernelGGL(cvt_f32_to_bf16, dim3((unsigned)(wElems / 2048)), block, 0, stream,
                       w_v, wvb, wElems);
    hipLaunchKernelGGL(routed_gemm_bf16_async, grid, block, 0, stream,
                       xb, mask, wvb, wtb, out);
  } else {
    hipLaunchKernelGGL(routed_dual_linear_fused, grid, block, 0, stream,
                       x, mask, w_v, w_t, out);
  }
}
